// LRTransformer_20194936226530
// MI455X (gfx1250) — matrix operand order repaired and checked by reading, not compiled, not run on hardware
//
// LRTransformer for MI455X (gfx1250): f16 WMMA GEMMs (16x16x32) for all dense
// layers, 32x64 tile per wave (8 WMMAs/K-step), global_prefetch for next K,
// fused BN/ReLU/bias/residual epilogues, kNN + neighborhood attention in LDS.
// wave32 fragment layouts per CDNA5 ISA 7.12.2.
#include <hip/hip_runtime.h>
#include <cstdint>
#include <cstddef>

typedef __attribute__((ext_vector_type(16))) _Float16 f16x16;
typedef __attribute__((ext_vector_type(8)))  _Float16 f16x8;
typedef __attribute__((ext_vector_type(8)))  float    v8f;

union F16Frag { f16x16 v; f16x8 h[2]; };
union AccFrag { v8f v; float f[8]; };

struct BN4 { const float *g, *b, *m, *v; };

// ---------------------------------------------------------------------------
// Conversion kernels: fp32 -> f16 with optional fused BN + ReLU, K padded to 32
// ---------------------------------------------------------------------------
__global__ void k_cvtA(const float* __restrict__ x, _Float16* __restrict__ y,
                       int M, int C, int Kp,
                       const float* __restrict__ g, const float* __restrict__ bb,
                       const float* __restrict__ mm, const float* __restrict__ vv,
                       int relu)
{
  int i = blockIdx.x * blockDim.x + threadIdx.x;
  if (i >= M * Kp) return;
  int m = i / Kp, k = i - m * Kp;
  float val = 0.f;
  if (k < C) {
    val = x[(size_t)m * C + k];
    if (g) val = (val - mm[k]) * (g[k] * rsqrtf(vv[k] + 1e-5f)) + bb[k];
    if (relu) val = fmaxf(val, 0.f);
  }
  y[i] = (_Float16)val;
}

__global__ void k_cvtW(const float* __restrict__ w, _Float16* __restrict__ y,
                       int C, int N, int Kp)
{
  int i = blockIdx.x * blockDim.x + threadIdx.x;
  if (i >= Kp * N) return;
  int k = i / N, n = i - k * N;
  y[i] = (k < C) ? (_Float16)w[(size_t)k * N + n] : (_Float16)0.f;
}

// ---------------------------------------------------------------------------
// WMMA GEMM: C[MxN] = A[MxKp] @ W[KpxN]  (f16 in, f32 acc)
// epilogue: (+bias[n]) -> (BN over n) -> (+residual[m][n]) -> (ReLU)
// One wave computes a 32x64 tile (2 A-frags x 4 B-frags = 8 WMMAs per K-step);
// 8 waves / 256-thread block. Next-K rows prefetched via global_prefetch.
// ---------------------------------------------------------------------------
__global__ __launch_bounds__(256) void k_gemm(
    const _Float16* __restrict__ A, const _Float16* __restrict__ W,
    float* __restrict__ Cm, int M, int N, int Kp,
    const float* __restrict__ bias,
    const float* __restrict__ g, const float* __restrict__ bb,
    const float* __restrict__ mm, const float* __restrict__ vv,
    const float* __restrict__ res, int relu)
{
  int lane = threadIdx.x & 31;
  int wid  = threadIdx.x >> 5;
  int tn64 = N >> 6;
  int gw   = blockIdx.x * 8 + wid;               // wave-uniform
  int total = (M >> 5) * tn64;
  if (gw >= total) return;                       // wave-uniform exit: EXEC stays all-1s
  int tm = gw / tn64, tn = gw - tm * tn64;

  int mrow0 = (tm << 5) + (lane & 15);           // A rows for frag 0
  int mrow1 = mrow0 + 16;                        // A rows for frag 1
  int khalf = (lane >> 4) << 3;                  // 0 or 8
  int krow  = (lane & 15) + ((lane >> 4) << 4);  // B-matrix row per lane

  AccFrag acc[2][4];
#pragma unroll
  for (int i = 0; i < 2; i++)
#pragma unroll
    for (int j = 0; j < 4; j++)
#pragma unroll
      for (int r = 0; r < 8; r++) acc[i][j].f[r] = 0.f;

  for (int k0 = 0; k0 < Kp; k0 += 32) {
    if (k0 + 32 < Kp) {
      // next-K prefetch (lowers to global_prefetch_b8; no LOADcnt cost)
      __builtin_prefetch(A + (size_t)mrow0 * Kp + k0 + 32 + khalf, 0, 1);
      __builtin_prefetch(A + (size_t)mrow1 * Kp + k0 + 32 + khalf, 0, 1);
      __builtin_prefetch(W + (size_t)(k0 + 32 + krow) * N + (tn << 6), 0, 1);
    }
    F16Frag a0, a1;
    const _Float16* ar0 = A + (size_t)mrow0 * Kp + k0 + khalf;
    const _Float16* ar1 = A + (size_t)mrow1 * Kp + k0 + khalf;
    a0.h[0] = *(const f16x8*)(ar0);
    a0.h[1] = *(const f16x8*)(ar0 + 16);
    a1.h[0] = *(const f16x8*)(ar1);
    a1.h[1] = *(const f16x8*)(ar1 + 16);
    const _Float16* br = W + (size_t)(k0 + krow) * N + (tn << 6);
#pragma unroll
    for (int j = 0; j < 4; j++) {
      F16Frag bf;
#pragma unroll
      for (int s = 0; s < 16; s++)
        bf.v[s] = W[(size_t)(k0 + khalf + (s & 7) + ((s >> 3) << 4)) * N + (tn << 6) + j * 16 + (lane & 15)];
      acc[0][j].v = __builtin_amdgcn_wmma_f32_16x16x32_f16(
          false, a0.v, false, bf.v, (short)0, acc[0][j].v, false, false);
      acc[1][j].v = __builtin_amdgcn_wmma_f32_16x16x32_f16(
          false, a1.v, false, bf.v, (short)0, acc[1][j].v, false, false);
    }
  }

  int ncol0 = (tn << 6);
#pragma unroll
  for (int j = 0; j < 4; j++) {
    int nc = ncol0 + j * 16 + (lane & 15);
    float bi = bias ? bias[nc] : 0.f;
    float sc = 1.f, sh = 0.f;
    if (g) { sc = g[nc] * rsqrtf(vv[nc] + 1e-5f); sh = bb[nc] - mm[nc] * sc; }
#pragma unroll
    for (int i = 0; i < 2; i++) {
      int m0 = (tm << 5) + (i << 4) + ((lane >> 4) << 3);
#pragma unroll
      for (int r = 0; r < 8; r++) {
        int mr = m0 + r;
        float val = acc[i][j].f[r] + bi;
        if (g) val = val * sc + sh;
        if (res) val += res[(size_t)mr * N + nc];
        if (relu) val = fmaxf(val, 0.f);
        Cm[(size_t)mr * N + nc] = val;
      }
    }
  }
}

// ---------------------------------------------------------------------------
// kNN: per query point, insertion-select 16 smallest squared distances.
// ---------------------------------------------------------------------------
__global__ void k_knn(const float* __restrict__ p, int* __restrict__ idx,
                      int Nn, int total)
{
  int t = blockIdx.x * blockDim.x + threadIdx.x;
  if (t >= total) return;
  int b = t / Nn;
  const float* pb = p + (size_t)b * Nn * 3;
  float px = p[(size_t)t * 3 + 0], py = p[(size_t)t * 3 + 1], pz = p[(size_t)t * 3 + 2];
  float bd[16]; int bi[16];
#pragma unroll
  for (int j = 0; j < 16; j++) { bd[j] = 3.0e38f; bi[j] = 0; }
  for (int j = 0; j < Nn; j++) {
    float dx = pb[j * 3 + 0] - px, dy = pb[j * 3 + 1] - py, dz = pb[j * 3 + 2] - pz;
    float d = dx * dx + dy * dy + dz * dz;
    if (d < bd[15]) {
      int q = 15;
      while (q > 0 && d < bd[q - 1]) { bd[q] = bd[q - 1]; bi[q] = bi[q - 1]; q--; }
      bd[q] = d; bi[q] = j;
    }
  }
  for (int j = 0; j < 16; j++) idx[(size_t)t * 16 + j] = bi[j];
}

// ---------------------------------------------------------------------------
// Neighborhood attention: one 256-thread block per (b, n) point.
// ---------------------------------------------------------------------------
struct AttnP {
  const float* p; const float* q; const float* k; const float* v; const int* idx;
  const float *Wp1, *bp1; BN4 pbn; const float *Wp2, *bp2;
  BN4 wbn1; const float *Ww1, *bw1; BN4 wbn2; const float *Ww2, *bw2;
  float* out; int Nn, pl, ms;
};

__global__ void k_attn(AttnP P)
{
  extern __shared__ float sm[];
  float* fb = sm;                      // [16*pl] relu(bn1(k - q + pr))
  float* gb = sm + 16 * P.pl;          // [16*pl] xv + pr
  float* t2 = gb + 16 * P.pl;          // [16*ms]
  float* ww = t2 + 16 * P.ms;          // [16*ms]
  __shared__ int   sidx[16];
  __shared__ float srel[48], shid[48];

  int tid = threadIdx.x;
  int bn_ = blockIdx.x;
  int b = bn_ / P.Nn, n = bn_ - b * P.Nn;

  if (tid < 16) sidx[tid] = P.idx[(size_t)bn_ * 16 + tid];
  __syncthreads();
  if (tid < 48) {
    int s = tid / 3, d = tid - s * 3;
    srel[tid] = P.p[((size_t)b * P.Nn + sidx[s]) * 3 + d] -
                P.p[((size_t)b * P.Nn + n) * 3 + d];
  }
  __syncthreads();
  if (tid < 48) {
    int s = tid / 3, d = tid - s * 3;
    float h = P.bp1[d];
    for (int e = 0; e < 3; e++) h += srel[s * 3 + e] * P.Wp1[e * 3 + d];
    h = (h - P.pbn.m[d]) * (P.pbn.g[d] * rsqrtf(P.pbn.v[d] + 1e-5f)) + P.pbn.b[d];
    shid[tid] = fmaxf(h, 0.f);
  }
  __syncthreads();

  const float* qrow = P.q + ((size_t)b * P.Nn + n) * P.pl;
  for (int i = tid; i < 16 * P.pl; i += blockDim.x) {
    int s = i / P.pl, c = i - s * P.pl;
    size_t nb = (size_t)b * P.Nn + sidx[s];
    float prv = P.bp2[c] + shid[s * 3 + 0] * P.Wp2[c]
                         + shid[s * 3 + 1] * P.Wp2[P.pl + c]
                         + shid[s * 3 + 2] * P.Wp2[2 * P.pl + c];
    gb[i] = P.v[nb * P.pl + c] + prv;
    float t = P.k[nb * P.pl + c] - qrow[c] + prv;
    t = (t - P.wbn1.m[c]) * (P.wbn1.g[c] * rsqrtf(P.wbn1.v[c] + 1e-5f)) + P.wbn1.b[c];
    fb[i] = fmaxf(t, 0.f);
  }
  __syncthreads();

  for (int i = tid; i < 16 * P.ms; i += blockDim.x) {
    int s = i / P.ms, mc = i - s * P.ms;
    float acc = P.bw1[mc];
    const float* fr = fb + (size_t)s * P.pl;
    for (int c = 0; c < P.pl; c++) acc += fr[c] * P.Ww1[(size_t)c * P.ms + mc];
    acc = (acc - P.wbn2.m[mc]) * (P.wbn2.g[mc] * rsqrtf(P.wbn2.v[mc] + 1e-5f)) + P.wbn2.b[mc];
    t2[i] = fmaxf(acc, 0.f);
  }
  __syncthreads();

  for (int i = tid; i < 16 * P.ms; i += blockDim.x) {
    int s = i / P.ms, mc = i - s * P.ms;
    float acc = P.bw2[mc];
    const float* tr = t2 + (size_t)s * P.ms;
    for (int c = 0; c < P.ms; c++) acc += tr[c] * P.Ww2[(size_t)c * P.ms + mc];
    ww[i] = acc;
  }
  __syncthreads();

  if (tid < P.ms) {
    float mx = -3.0e38f;
    for (int s = 0; s < 16; s++) mx = fmaxf(mx, ww[s * P.ms + tid]);
    float e[16], sum = 0.f;
    for (int s = 0; s < 16; s++) { e[s] = __expf(ww[s * P.ms + tid] - mx); sum += e[s]; }
    float inv = 1.f / sum;
    for (int s = 0; s < 16; s++) ww[s * P.ms + tid] = e[s] * inv;
  }
  __syncthreads();

  float* orow = P.out + ((size_t)b * P.Nn + n) * P.pl;
  for (int c = tid; c < P.pl; c += blockDim.x) {
    int mc = c % P.ms;
    float acc = 0.f;
    for (int s = 0; s < 16; s++) acc += gb[(size_t)s * P.pl + c] * ww[s * P.ms + mc];
    orow[c] = acc;
  }
}

// ---------------------------------------------------------------------------
// Small elementwise helpers
// ---------------------------------------------------------------------------
__global__ void k_copyp(const float* __restrict__ x, float* __restrict__ p,
                        int C, int total)
{
  int i = blockIdx.x * blockDim.x + threadIdx.x;
  if (i >= total * 3) return;
  int bn = i / 3, d = i - bn * 3;
  p[i] = x[(size_t)bn * C + d];
}

__global__ void k_mean(const float* __restrict__ x, float* __restrict__ pool,
                       int Nn, int C, int poolStride, int off)
{
  int i = blockIdx.x * blockDim.x + threadIdx.x;
  if (i >= 2 * C) return;
  int b = i / C, c = i - b * C;
  const float* xr = x + (size_t)b * Nn * C;
  float s = 0.f;
  for (int n = 0; n < Nn; n++) s += xr[(size_t)n * C + c];
  pool[(size_t)b * poolStride + off + c] = s * (1.f / Nn);
}

__global__ void k_build3(const float* __restrict__ pool, const float* __restrict__ r,
                         float* __restrict__ out, int Nn)
{
  size_t i = (size_t)blockIdx.x * blockDim.x + threadIdx.x;
  size_t total = (size_t)2 * Nn * 2176;
  if (i >= total) return;
  int c = (int)(i % 2176);
  size_t bn = i / 2176;
  int b = (int)(bn / Nn);
  out[i] = (c < 2048) ? pool[(size_t)b * 2048 + c] : r[bn * 128 + (c - 2048)];
}

__global__ void k_cls(const float* __restrict__ h, const float* __restrict__ Wc,
                      const float* __restrict__ bc, float* __restrict__ out, int total)
{
  int i = blockIdx.x * blockDim.x + threadIdx.x;
  if (i >= total * 2) return;
  int o = i & 1, bn = i >> 1;
  const float* hr = h + (size_t)bn * 256;
  float s = bc[o];
  for (int c = 0; c < 256; c++) s += hr[c] * Wc[c * 2 + o];
  out[i] = s;
}

// ---------------------------------------------------------------------------
// Host-side parameter walking and orchestration
// ---------------------------------------------------------------------------
struct BlockP {
  int cin, pl, ms;
  const float* W1; BN4 bn1;
  const float *Wq, *bq, *Wk, *bk, *Wv, *bv;
  const float *Wp1, *bp1; BN4 pbn; const float *Wp2, *bp2;
  BN4 wbn1; const float *Ww1, *bw1; BN4 wbn2; const float *Ww2, *bw2;
  BN4 bn2; const float* W3; BN4 bn3;
};
struct HeadP { const float *W1, *b1; BN4 bn; const float *Wc, *bc; };

struct Cursor {
  void* const* a; int i;
  const float* f() { return (const float*)a[i++]; }
};
static BN4 rdBN(Cursor& c) { BN4 r; r.g = c.f(); r.b = c.f(); r.m = c.f(); r.v = c.f(); return r; }
static BlockP rdBlock(Cursor& c, int cin, int pl) {
  BlockP p; p.cin = cin; p.pl = pl; p.ms = pl / 32;
  p.W1 = c.f(); p.bn1 = rdBN(c);
  p.Wq = c.f(); p.bq = c.f(); p.Wk = c.f(); p.bk = c.f(); p.Wv = c.f(); p.bv = c.f();
  p.Wp1 = c.f(); p.bp1 = c.f(); p.pbn = rdBN(c); p.Wp2 = c.f(); p.bp2 = c.f();
  p.wbn1 = rdBN(c); p.Ww1 = c.f(); p.bw1 = c.f();
  p.wbn2 = rdBN(c); p.Ww2 = c.f(); p.bw2 = c.f();
  p.bn2 = rdBN(c); p.W3 = c.f(); p.bn3 = rdBN(c);
  return p;
}
static HeadP rdHead(Cursor& c) {
  HeadP h; h.W1 = c.f(); h.b1 = c.f(); h.bn = rdBN(c); h.Wc = c.f(); h.bc = c.f();
  return h;
}

struct WS {
  float *p, *x1, *qb, *kb, *vb, *attn, *bufA, *bufB, *r1, *a1, *segout, *pool, *segin, *headh;
  int* idx;
  _Float16 *abf, *wbf;
};

static void cvtA(hipStream_t st, const float* x, _Float16* y, int M, int C, int Kp,
                 const BN4* bn, int relu) {
  int total = M * Kp;
  k_cvtA<<<(total + 255) / 256, 256, 0, st>>>(x, y, M, C, Kp,
      bn ? bn->g : nullptr, bn ? bn->b : nullptr,
      bn ? bn->m : nullptr, bn ? bn->v : nullptr, relu);
}
static void cvtW(hipStream_t st, const float* w, _Float16* y, int C, int N, int Kp) {
  int total = Kp * N;
  k_cvtW<<<(total + 255) / 256, 256, 0, st>>>(w, y, C, N, Kp);
}
static void gemm(hipStream_t st, const _Float16* A, const _Float16* W, float* C,
                 int M, int N, int Kp, const float* bias, const BN4* bn,
                 const float* res, int relu) {
  int waves = (M / 32) * (N / 64);
  k_gemm<<<(waves + 7) / 8, 256, 0, st>>>(A, W, C, M, N, Kp, bias,
      bn ? bn->g : nullptr, bn ? bn->b : nullptr,
      bn ? bn->m : nullptr, bn ? bn->v : nullptr, res, relu);
}

static void run_block(hipStream_t st, const WS& ws, const BlockP& P,
                      const float* xin, int cin, float* xout) {
  const int M = 1024, Nn = 512;
  int pl = P.pl;
  int Kp0 = (cin + 31) & ~31;

  // x1 = relu(bn1(x @ W1))      (identity for residual)
  cvtA(st, xin, ws.abf, M, cin, Kp0, nullptr, 0);
  cvtW(st, P.W1, ws.wbf, cin, pl, Kp0);
  gemm(st, ws.abf, ws.wbf, ws.x1, M, pl, Kp0, nullptr, &P.bn1, nullptr, 1);

  // q, k, v projections (bias only)
  cvtA(st, ws.x1, ws.abf, M, pl, pl, nullptr, 0);
  cvtW(st, P.Wq, ws.wbf, pl, pl, pl);
  gemm(st, ws.abf, ws.wbf, ws.qb, M, pl, pl, P.bq, nullptr, nullptr, 0);
  cvtW(st, P.Wk, ws.wbf, pl, pl, pl);
  gemm(st, ws.abf, ws.wbf, ws.kb, M, pl, pl, P.bk, nullptr, nullptr, 0);
  cvtW(st, P.Wv, ws.wbf, pl, pl, pl);
  gemm(st, ws.abf, ws.wbf, ws.vb, M, pl, pl, P.bv, nullptr, nullptr, 0);

  AttnP ap;
  ap.p = ws.p; ap.q = ws.qb; ap.k = ws.kb; ap.v = ws.vb; ap.idx = ws.idx;
  ap.Wp1 = P.Wp1; ap.bp1 = P.bp1; ap.pbn = P.pbn; ap.Wp2 = P.Wp2; ap.bp2 = P.bp2;
  ap.wbn1 = P.wbn1; ap.Ww1 = P.Ww1; ap.bw1 = P.bw1;
  ap.wbn2 = P.wbn2; ap.Ww2 = P.Ww2; ap.bw2 = P.bw2;
  ap.out = ws.attn; ap.Nn = Nn; ap.pl = pl; ap.ms = P.ms;
  size_t sh = (size_t)(32 * pl + 32 * P.ms) * sizeof(float);
  k_attn<<<M, 256, sh, st>>>(ap);

  // out = relu(bn3(relu(bn2(attn)) @ W3) + identity)
  cvtA(st, ws.attn, ws.abf, M, pl, pl, &P.bn2, 1);
  cvtW(st, P.W3, ws.wbf, pl, pl, pl);
  gemm(st, ws.abf, ws.wbf, xout, M, pl, pl, nullptr, &P.bn3, ws.x1, 1);
}

static void run_seg(hipStream_t st, const WS& ws, const BlockP* blk, int nb,
                    const float* input, int cin0, float* final_out,
                    const HeadP* head, float* cls_out) {
  const int M = 1024, Nn = 512;
  k_copyp<<<(M * 3 + 255) / 256, 256, 0, st>>>(input, ws.p, cin0, M);
  k_knn<<<(M + 255) / 256, 256, 0, st>>>(ws.p, ws.idx, Nn, M);
  const float* cur = input;
  int cin = cin0;
  for (int i = 0; i < nb; i++) {
    float* xout = (i == nb - 1) ? final_out : ((i & 1) ? ws.bufB : ws.bufA);
    run_block(st, ws, blk[i], cur, cin, xout);
    cur = xout; cin = blk[i].pl;
  }
  if (head) {
    cvtA(st, final_out, ws.abf, M, 128, 128, nullptr, 0);
    cvtW(st, head->W1, ws.wbf, 128, 256, 128);
    gemm(st, ws.abf, ws.wbf, ws.headh, M, 256, 128, head->b1, &head->bn, nullptr, 1);
    k_cls<<<(M * 2 + 255) / 256, 256, 0, st>>>(ws.headh, head->Wc, head->bc, cls_out, M);
  }
}

extern "C" void kernel_launch(void* const* d_in, const int* in_sizes, int n_in,
                              void* d_out, int out_size, void* d_ws, size_t ws_size,
                              hipStream_t stream)
{
  (void)in_sizes; (void)n_in; (void)out_size; (void)ws_size;
  const float* inlier = (const float*)d_in[0];
  const float* neigh  = (const float*)d_in[1];

  Cursor c{d_in, 2};
  static const int cfg1[2][2] = {{13, 128}, {128, 128}};
  static const int cfg2[4][2] = {{128, 128}, {128, 256}, {256, 512}, {512, 1024}};
  static const int cfg3[3][2] = {{2176, 512}, {512, 256}, {256, 128}};
  BlockP b1i[2], b1n[2], b2i[4], b2n[4], b3i[3], b3n[3];
  for (int i = 0; i < 2; i++) b1i[i] = rdBlock(c, cfg1[i][0], cfg1[i][1]);
  for (int i = 0; i < 2; i++) b1n[i] = rdBlock(c, cfg1[i][0], cfg1[i][1]);
  for (int i = 0; i < 4; i++) b2i[i] = rdBlock(c, cfg2[i][0], cfg2[i][1]);
  for (int i = 0; i < 4; i++) b2n[i] = rdBlock(c, cfg2[i][0], cfg2[i][1]);
  for (int i = 0; i < 3; i++) b3i[i] = rdBlock(c, cfg3[i][0], cfg3[i][1]);
  for (int i = 0; i < 3; i++) b3n[i] = rdBlock(c, cfg3[i][0], cfg3[i][1]);
  HeadP hi = rdHead(c), hn = rdHead(c);

  // workspace carve-out
  char* base = (char*)d_ws; size_t off = 0;
  auto alignup = [&]() { off = (off + 255) & ~(size_t)255; };
  auto af = [&](size_t n) -> float* { alignup(); float* r = (float*)(base + off); off += n * sizeof(float); return r; };
  WS ws;
  ws.p      = af((size_t)1024 * 3);
  ws.x1     = af((size_t)1024 * 1024);
  ws.qb     = af((size_t)1024 * 1024);
  ws.kb     = af((size_t)1024 * 1024);
  ws.vb     = af((size_t)1024 * 1024);
  ws.attn   = af((size_t)1024 * 1024);
  ws.bufA   = af((size_t)1024 * 1024);
  ws.bufB   = af((size_t)1024 * 1024);
  ws.r1     = af((size_t)1024 * 128);
  ws.a1     = af((size_t)1024 * 128);
  ws.segout = af((size_t)1024 * 1024);
  ws.pool   = af((size_t)2 * 2048);
  ws.segin  = af((size_t)1024 * 2176);
  ws.headh  = af((size_t)1024 * 256);
  alignup(); ws.idx = (int*)(base + off);       off += (size_t)1024 * 16 * sizeof(int);
  alignup(); ws.abf = (_Float16*)(base + off);  off += (size_t)1024 * 2176 * sizeof(_Float16);
  alignup(); ws.wbf = (_Float16*)(base + off);  off += (size_t)2176 * 1024 * sizeof(_Float16);

  float* outF = (float*)d_out;

  // segment 1 (both branches)
  run_seg(stream, ws, b1i, 2, inlier, 13, ws.r1, nullptr, nullptr);
  run_seg(stream, ws, b1n, 2, neigh, 13, ws.a1, nullptr, nullptr);

  // segment 2 + means
  run_seg(stream, ws, b2i, 4, ws.r1, 128, ws.segout, nullptr, nullptr);
  k_mean<<<(2 * 1024 + 255) / 256, 256, 0, stream>>>(ws.segout, ws.pool, 512, 1024, 2048, 0);
  run_seg(stream, ws, b2n, 4, ws.a1, 128, ws.segout, nullptr, nullptr);
  k_mean<<<(2 * 1024 + 255) / 256, 256, 0, stream>>>(ws.segout, ws.pool, 512, 1024, 2048, 1024);

  // segment 3 + heads
  {
    size_t total = (size_t)2 * 512 * 2176;
    k_build3<<<(unsigned)((total + 255) / 256), 256, 0, stream>>>(ws.pool, ws.r1, ws.segin, 512);
    run_seg(stream, ws, b3i, 3, ws.segin, 2176, ws.segout, &hi, outF);
    k_build3<<<(unsigned)((total + 255) / 256), 256, 0, stream>>>(ws.pool, ws.a1, ws.segin, 512);
    run_seg(stream, ws, b3n, 3, ws.segin, 2176, ws.segout, &hn, outF + 2048);
  }
}
